// GDGF_86620900426013
// MI455X (gfx1250) — matrix operand order repaired and checked by reading, not compiled, not run on hardware
//
#include <hip/hip_runtime.h>
#include <hip/hip_bf16.h>

#define Bsz 8
#define Nn  2048
#define Dd  256

typedef _Float16 half_t;
typedef __attribute__((ext_vector_type(16))) _Float16 v16h;
typedef __attribute__((ext_vector_type(8)))  _Float16 v8h;
typedef __attribute__((ext_vector_type(8)))  float    v8f;

// ---- WMMA operand loaders (f16, per CDNA5 ISA §7.12.2 layouts) ----
// A (16x32, MxK): lane l (M = l%16), grp = l/16:
//   a[i], i=0..7  -> K = klo + grp*8 + i          (contiguous 16B)
//   a[i], i=8..15 -> K = klo + 16 + grp*8 + (i-8) (contiguous 16B)
__device__ __forceinline__ v16h load_a_f16(const half_t* rowptr, int klo, int grp) {
    const v8h lo = *(const v8h*)(rowptr + klo + grp * 8);
    const v8h hi = *(const v8h*)(rowptr + klo + 16 + grp * 8);
    v16h a;
#pragma unroll
    for (int i = 0; i < 8; ++i) { a[i] = lo[i]; a[i + 8] = hi[i]; }
    return a;
}
// B (32x16, KxN): lane l (N = l%16), grp = l/16:
//   b[i], i=0..7  -> K = klo + grp*8 + i          (contiguous 16B)
//   b[i], i=8..15 -> K = klo + 16 + grp*8 + (i-8) (contiguous 16B)
__device__ __forceinline__ v16h load_b_f16(const half_t* colptr, int klo, int grp) {
    const v8h lo = *(const v8h*)(colptr + klo + grp * 8);
    const v8h hi = *(const v8h*)(colptr + klo + 16 + grp * 8);
    v16h b;
#pragma unroll
    for (int i = 0; i < 8; ++i) { b[i] = lo[i]; b[i + 8] = hi[i]; }
    return b;
}

// ---- K0: transpose + convert W_fp (f32 [k][e]) -> Wt (f16 [e][k]) ----
__global__ __launch_bounds__(256) void wcvt_kernel(
    const float* __restrict__ W, half_t* __restrict__ Wt)
{
    const int e = blockIdx.x, k = threadIdx.x;
    Wt[(size_t)e * Dd + k] = (half_t)W[(size_t)k * Dd + e];
}

// ---- K1: proj = x @ W_fp + b_fp  (f16 WMMA, f32 accumulate) ----
__global__ __launch_bounds__(128) void proj_kernel(
    const float* __restrict__ x, const half_t* __restrict__ Wth,
    const float* __restrict__ bfp, float* __restrict__ proj)
{
    __shared__ half_t sA[16 * Dd];
    const int r0   = blockIdx.x * 16;              // global row in [0, B*N)
    const int t    = threadIdx.x;
    const int wave = t >> 5, lane = t & 31;
    const int grp  = lane >> 4, ln16 = lane & 15;
    const int col  = blockIdx.y * 64 + wave * 16 + ln16;

    // Stage the 16x256 x row-block into LDS as f16 (once per WG)
    {
        const int row = t >> 3, seg = t & 7;
        const float* src = x + (size_t)(r0 + row) * Dd + seg * 32;
        half_t* dst = &sA[row * Dd + seg * 32];
#pragma unroll
        for (int j = 0; j < 32; ++j) dst[j] = (half_t)src[j];
    }
    __syncthreads();

    v16h areg[8];
#pragma unroll
    for (int ks = 0; ks < 8; ++ks) areg[ks] = load_a_f16(&sA[ln16 * Dd], ks * 32, grp);

    const half_t* wcol = Wth + (size_t)col * Dd;
    v8f c = {};
#pragma unroll
    for (int ks = 0; ks < 8; ++ks) {
        const v16h bb = load_b_f16(wcol, ks * 32, grp);
        c = __builtin_amdgcn_wmma_f32_16x16x32_f16(false, areg[ks], false, bb, (short)0, c, false, false);
    }
    const float bias = bfp[col];
#pragma unroll
    for (int r = 0; r < 8; ++r)
        proj[(size_t)(r0 + grp * 8 + r) * Dd + col] = c[r] + bias;
}

// ---- K2: per-row L2 norm -> xn (f16); s = sigmoid(proj . w_s + b_s) ----
__global__ __launch_bounds__(256) void rownorm_kernel(
    const float* __restrict__ proj, const float* __restrict__ wsv,
    const float* __restrict__ bs, half_t* __restrict__ xnh, float* __restrict__ s)
{
    const int t    = threadIdx.x;
    const int row  = blockIdx.x * 8 + (t >> 5);
    const int lane = t & 31;
    const float* pr = proj + (size_t)row * Dd;
    float sumsq = 0.f, dot = 0.f, vals[8];
#pragma unroll
    for (int j = 0; j < 8; ++j) {
        const float v = pr[lane + j * 32];
        vals[j] = v;
        sumsq += v * v;
        dot   += v * wsv[lane + j * 32];
    }
#pragma unroll
    for (int off = 16; off > 0; off >>= 1) {
        sumsq += __shfl_xor(sumsq, off, 32);
        dot   += __shfl_xor(dot,   off, 32);
    }
    const float invn = 1.0f / fmaxf(sqrtf(sumsq), 1e-12f);
    half_t* xr = xnh + (size_t)row * Dd;
#pragma unroll
    for (int j = 0; j < 8; ++j) xr[lane + j * 32] = (half_t)(vals[j] * invn);
    if (lane == 0) s[row] = 1.0f / (1.0f + expf(-(dot + bs[0])));
}

// ---- K2b: partial column sums of x over N (deterministic 2-stage mean) ----
__global__ __launch_bounds__(256) void xsum_kernel(
    const float* __restrict__ x, float* __restrict__ xpart)
{
    const int b = blockIdx.x >> 5, chunk = blockIdx.x & 31, d = threadIdx.x;
    const float* xb = x + ((size_t)b * Nn + (size_t)chunk * 64) * Dd;
    float acc = 0.f;
    for (int n = 0; n < 64; ++n) acc += xb[(size_t)n * Dd + d];
    xpart[(b * 32 + chunk) * Dd + d] = acc;
}

// ---- K3: sim = xn @ xn^T (f16 WMMA), adj epilogue + row-degree ----
// 4 waves/WG; each wave owns TWO independent 16x16 column tiles per
// iteration (two accumulator chains interleaved for WMMA ILP).
__global__ __launch_bounds__(128) void sim_adj_kernel(
    const half_t* __restrict__ xnh, const float* __restrict__ s,
    float* __restrict__ adj, float* __restrict__ invdeg)
{
    const int wg = blockIdx.x;
    const int b  = wg >> 7;              // N/16 = 128 row-blocks per batch
    const int r0 = (wg & 127) << 4;
    const int t  = threadIdx.x;
    const int wave = t >> 5, lane = t & 31;
    const int grp = lane >> 4, ln16 = lane & 15;

    const half_t* xb   = xnh + (size_t)b * Nn * Dd;
    const float*  sb   = s + b * Nn;
    float*        adjb = adj + (size_t)b * Nn * Nn;

    // A operand (rows r0..r0+15, full K=256) held in registers, reused 32x per wave
    v16h areg[8];
    const half_t* arow = xb + (size_t)(r0 + ln16) * Dd;
#pragma unroll
    for (int ks = 0; ks < 8; ++ks) areg[ks] = load_a_f16(arow, ks * 32, grp);

    float srow[8], degacc[8];
#pragma unroll
    for (int r = 0; r < 8; ++r) { srow[r] = sb[r0 + grp * 8 + r]; degacc[r] = 0.f; }

    float* rowbase = adjb + (size_t)(r0 + grp * 8) * Nn;

    for (int it = 0; it < Nn / 128; ++it) {
        const int col0 = it * 128 + wave * 32 + ln16;
        const int col1 = col0 + 16;
        const half_t* brow0 = xb + (size_t)col0 * Dd;
        const half_t* brow1 = xb + (size_t)col1 * Dd;
        v8f c0 = {}, c1 = {};
#pragma unroll
        for (int ks = 0; ks < 8; ++ks) {
            const v16h b0 = load_b_f16(brow0, ks * 32, grp);
            const v16h b1 = load_b_f16(brow1, ks * 32, grp);
            c0 = __builtin_amdgcn_wmma_f32_16x16x32_f16(false, areg[ks], false, b0, (short)0, c0, false, false);
            c1 = __builtin_amdgcn_wmma_f32_16x16x32_f16(false, areg[ks], false, b1, (short)0, c1, false, false);
        }
        const float scol0 = sb[col0];
        const float scol1 = sb[col1];
#pragma unroll
        for (int r = 0; r < 8; ++r) {
            float sc0 = fminf(fmaxf(0.5f * (srow[r] + scol0), 1e-6f), 1.0f);
            float sc1 = fminf(fmaxf(0.5f * (srow[r] + scol1), 1e-6f), 1.0f);
            const float av0 = fmaxf(sc0 * c0[r], 1e-6f);
            const float av1 = fmaxf(sc1 * c1[r], 1e-6f);
            rowbase[(size_t)r * Nn + col0] = av0;
            rowbase[(size_t)r * Nn + col1] = av1;
            degacc[r] += av0 + av1;
        }
    }

    __shared__ float degLds[128][8];
#pragma unroll
    for (int r = 0; r < 8; ++r) degLds[t][r] = degacc[r];
    __syncthreads();
    if (t < 16) {
        const int g = t >> 3, rr = t & 7;
        float sum = 0.f;
        for (int w = 0; w < 4; ++w)
            for (int li = 0; li < 16; ++li)
                sum += degLds[w * 32 + g * 16 + li][rr];
        invdeg[b * Nn + r0 + t] = 1.0f / fmaxf(sum, 1e-6f);
    }
}

// ---- K4: segmented column sums of adj * invdeg (L2-resident adj) ----
__global__ __launch_bounds__(256) void colsum_kernel(
    const float* __restrict__ adj, const float* __restrict__ invdeg,
    float* __restrict__ colwp)
{
    const int wg  = blockIdx.x;
    const int b   = wg >> 5;
    const int seg = (wg >> 3) & 3;
    const int col = (wg & 7) * 256 + threadIdx.x;
    const float* ab  = adj + (size_t)b * Nn * Nn;
    const float* idg = invdeg + b * Nn;
    float acc = 0.f;
    for (int n = seg * 512; n < seg * 512 + 512; ++n)
        acc += ab[(size_t)n * Nn + col] * idg[n];
    colwp[(b * 4 + seg) * Nn + col] = acc;
}

// ---- K5: t[b,d] = colw . proj[:, d] ----
__global__ __launch_bounds__(256) void tvec_kernel(
    const float* __restrict__ colwp, const float* __restrict__ proj,
    float* __restrict__ tv)
{
    const int b = blockIdx.x, d = threadIdx.x;
    const float* pb = proj + (size_t)b * Nn * Dd;
    const float* cw = colwp + (size_t)b * 4 * Nn;
    float acc = 0.f;
    for (int m = 0; m < Nn; ++m) {
        const float w = cw[m] + cw[Nn + m] + cw[2 * Nn + m] + cw[3 * Nn + m];
        acc += w * pb[(size_t)m * Dd + d];
    }
    tv[b * Dd + d] = acc;
}

// ---- K6: matvecs + global add + LayerNorm ----
__global__ __launch_bounds__(256) void final_kernel(
    const float* __restrict__ tv, const float* __restrict__ xpart,
    const float* __restrict__ Wrp, const float* __restrict__ brp,
    const float* __restrict__ Wgp, const float* __restrict__ bgp,
    const float* __restrict__ gamma, const float* __restrict__ beta,
    float* __restrict__ out)
{
    __shared__ float rm[Dd];
    __shared__ float u[Dd];
    __shared__ float red[256];
    const int b = blockIdx.x, d = threadIdx.x;
    float xm = 0.f;
    for (int c = 0; c < 32; ++c) xm += xpart[(b * 32 + c) * Dd + d];
    xm *= (1.0f / Nn);
    rm[d] = tv[b * Dd + d] * (1.0f / Nn);
    __syncthreads();
    float acc = brp[d];
    for (int k = 0; k < Dd; ++k) acc += rm[k] * Wrp[k * Dd + d];
    u[d] = acc;
    __syncthreads();
    float g = bgp[d];
    for (int k = 0; k < Dd; ++k) g += u[k] * Wgp[k * Dd + d];
    const float h = g + xm;
    red[d] = h; __syncthreads();
    for (int off = 128; off > 0; off >>= 1) { if (d < off) red[d] += red[d + off]; __syncthreads(); }
    const float mu = red[0] * (1.0f / Dd); __syncthreads();
    const float dv = h - mu;
    red[d] = dv * dv; __syncthreads();
    for (int off = 128; off > 0; off >>= 1) { if (d < off) red[d] += red[d + off]; __syncthreads(); }
    const float var = red[0] * (1.0f / Dd);
    out[b * Dd + d] = dv * rsqrtf(var + 1e-5f) * gamma[d] + beta[d];
}

extern "C" void kernel_launch(void* const* d_in, const int* in_sizes, int n_in,
                              void* d_out, int out_size, void* d_ws, size_t ws_size,
                              hipStream_t stream)
{
    (void)in_sizes; (void)n_in; (void)out_size; (void)ws_size;
    const float* x    = (const float*)d_in[0];
    const float* Wfp  = (const float*)d_in[1];
    const float* bfp  = (const float*)d_in[2];
    const float* wsv  = (const float*)d_in[3];
    const float* bs   = (const float*)d_in[4];
    const float* Wrp  = (const float*)d_in[5];
    const float* brp  = (const float*)d_in[6];
    const float* Wgp  = (const float*)d_in[7];
    const float* bgp  = (const float*)d_in[8];
    const float* gam  = (const float*)d_in[9];
    const float* bet  = (const float*)d_in[10];

    float* out_ln = (float*)d_out;                 // [B,1,D] first
    float* adj    = (float*)d_out + Bsz * Dd;      // then [B,N,N]

    char* ws = (char*)d_ws;
    float*  proj   = (float*)ws;  ws += (size_t)Bsz * Nn * Dd * 4;
    half_t* xnh    = (half_t*)ws; ws += (size_t)Bsz * Nn * Dd * 2;
    half_t* Wth    = (half_t*)ws; ws += (size_t)Dd * Dd * 2;
    float*  s      = (float*)ws;  ws += (size_t)Bsz * Nn * 4;
    float*  invdeg = (float*)ws;  ws += (size_t)Bsz * Nn * 4;
    float*  xpart  = (float*)ws;  ws += (size_t)Bsz * 32 * Dd * 4;
    float*  colwp  = (float*)ws;  ws += (size_t)Bsz * 4 * Nn * 4;
    float*  tv     = (float*)ws;  ws += (size_t)Bsz * Dd * 4;

    wcvt_kernel   <<<Dd, Dd, 0, stream>>>(Wfp, Wth);
    proj_kernel   <<<dim3(Bsz * Nn / 16, Dd / 64), 128, 0, stream>>>(x, Wth, bfp, proj);
    rownorm_kernel<<<Bsz * Nn / 8, 256, 0, stream>>>(proj, wsv, bs, xnh, s);
    xsum_kernel   <<<Bsz * 32, 256, 0, stream>>>(x, xpart);
    sim_adj_kernel<<<Bsz * (Nn / 16), 128, 0, stream>>>(xnh, s, adj, invdeg);
    colsum_kernel <<<256, 256, 0, stream>>>(adj, invdeg, colwp);
    tvec_kernel   <<<Bsz, 256, 0, stream>>>(colwp, proj, tv);
    final_kernel  <<<Bsz, 256, 0, stream>>>(tv, xpart, Wrp, brp, Wgp, bgp, gam, bet, out_ln);
}
